// EquivariantBinaryClassificationNoGraphScalar_39513699123758
// MI455X (gfx1250) — compile-verified, matrix operand order checked by reading, not run on hardware
//
#include <hip/hip_runtime.h>
#include <math.h>

// CDNA5 / gfx1250. Wave32. WMMA via __builtin_amdgcn_wmma_f32_16x16x32_f16.
typedef __attribute__((ext_vector_type(16))) _Float16 v16h;
typedef __attribute__((ext_vector_type(8)))  float    v8f;

#define LN_EPS 1e-5f

// ---------------------------------------------------------------------------
// Kernel 1: s[b, c] = sum_n x[b, n, c].   (bandwidth-bound: streams 1 GiB)
// grid = B blocks, block = C/4 threads (256 for C=1024), float4 per thread.
// Each 'n' step the block reads a contiguous C*4 = 4KB span: fully coalesced
// global_load_b128s; 8 waves/block * B blocks gives ample memory parallelism.
// ---------------------------------------------------------------------------
__global__ void __launch_bounds__(256)
reduce_n_kernel(const float* __restrict__ x, float* __restrict__ s,
                int N, int C) {
    const int b  = blockIdx.x;
    const int c4 = threadIdx.x * 4;
    const float* xp = x + (size_t)b * (size_t)N * (size_t)C + c4;
    float4 acc = make_float4(0.f, 0.f, 0.f, 0.f);
#pragma unroll 4
    for (int n = 0; n < N; ++n) {
        const float4 v = *reinterpret_cast<const float4*>(xp + (size_t)n * C);
        acc.x += v.x; acc.y += v.y; acc.z += v.z; acc.w += v.w;
    }
    *reinterpret_cast<float4*>(s + (size_t)b * C + c4) = acc;
}

// ---------------------------------------------------------------------------
// Kernel 2: per 16 batches: LayerNorm(s) -> h, then logits = h . W via WMMA,
// out = sigmoid(logits + bias).  grid = B/16 blocks, 256 threads.
//
// WMMA plan (f16 in, f32 accum), D = A x B + C with 16x16x32:
//   A[m, k] = h[b0+m, c0+k]          (16 batches x 32 channels per step)
//   B[k, n] = W[c0+k]  (same for all n -> every column of D is the logit)
// C = 1024 -> 32 WMMA steps, chained f32 accumulators.
//
// A-fragment element mapping (16-bit A 16x32, ISA 7.12.2): lane l = m + 16*g,
// half-element e in the 8-VGPR (v16h) operand corresponds to
//   kc = (e & 7) | (g << 3) | ((e & 8) << 1)      (kc = K within the tile)
// We write h into LDS already in this order so stage C is 32B/lane loads.
// ---------------------------------------------------------------------------
__device__ __forceinline__ int frag_e_of_kc(int kc) {
    // e for A/B fragment given K-in-tile; g = (kc>>3)&1 handled by lane/row.
    return (kc & 7) | ((kc & 16) >> 1);
}

__global__ void __launch_bounds__(256)
ln_head_kernel(const float* __restrict__ s,
               const float* __restrict__ ln_w, const float* __restrict__ ln_b,
               const float* __restrict__ W, const float* __restrict__ bias,
               float* __restrict__ out, int C) {
    // C is assumed 1024 (static LDS sizing); B multiple of 16.
    __shared__ __align__(32) _Float16 h_frag[16 * 1024]; // 32KB: A fragments
    __shared__ __align__(32) _Float16 w_frag[64 * 16];   // 2KB : B fragments
    __shared__ float red_sum[256];
    __shared__ float red_sq[256];
    __shared__ float mu_s[16];
    __shared__ float rs_s[16];

    const int tid = threadIdx.x;
    const int b0  = blockIdx.x * 16;

    // ---- Stage A: per-batch mean / rstd over C channels -------------------
    {
        const int m   = tid >> 4;    // batch row 0..15
        const int sub = tid & 15;    // 16 threads per row
        const float* sp = s + (size_t)(b0 + m) * C;
        float sum = 0.f, sq = 0.f;
        for (int c = sub; c < C; c += 16) {
            float v = sp[c];
            sum += v; sq += v * v;
        }
        red_sum[tid] = sum;
        red_sq[tid]  = sq;
    }
    __syncthreads();
    if (tid < 16) {
        float sum = 0.f, sq = 0.f;
#pragma unroll
        for (int k = 0; k < 16; ++k) {
            sum += red_sum[tid * 16 + k];
            sq  += red_sq[tid * 16 + k];
        }
        const float inv_c = 1.0f / (float)C;
        const float mu  = sum * inv_c;
        const float var = fmaxf(sq * inv_c - mu * mu, 0.0f);
        mu_s[tid] = mu;
        rs_s[tid] = rsqrtf(var + LN_EPS);
    }
    // ---- Stage A': W -> B-fragment layout in LDS --------------------------
    // Each channel c maps to a unique (tile, g, e) slot; 1024 slots total.
    for (int c = tid; c < C; c += 256) {
        const int tile = c >> 5;
        const int kc   = c & 31;
        const int g    = (kc >> 3) & 1;
        const int e    = frag_e_of_kc(kc);
        w_frag[(tile * 2 + g) * 16 + e] = (_Float16)W[c];
    }
    __syncthreads();

    // ---- Stage B: h = (s - mu) * rstd * ln_w + ln_b -> A-fragment LDS -----
    const int total = 16 * C;
    for (int i = tid; i < total; i += 256) {
        const int c = i & (C - 1);
        const int m = i >> 10;                 // C == 1024
        const float v  = s[(size_t)(b0 + m) * C + c];
        const float hv = (v - mu_s[m]) * rs_s[m] * ln_w[c] + ln_b[c];
        const int tile = c >> 5;
        const int kc   = c & 31;
        const int g    = (kc >> 3) & 1;
        const int e    = frag_e_of_kc(kc);
        const int lane = m + (g << 4);
        h_frag[(tile * 32 + lane) * 16 + e] = (_Float16)hv;
    }
    __syncthreads();

    // ---- Stage C: wave 0 does the matvec on the matrix engine -------------
    if (tid < 32) {
        const int l = tid;
        const int g = l >> 4;
        const int ntiles = C >> 5;             // 32
        v8f acc0 = {};
        v8f acc1 = {};
        for (int t = 0; t < ntiles; t += 2) {
            v16h a0 = *reinterpret_cast<const v16h*>(&h_frag[(t * 32 + l) * 16]);
            v16h bf0 = *reinterpret_cast<const v16h*>(&w_frag[(t * 2 + g) * 16]);
            acc0 = __builtin_amdgcn_wmma_f32_16x16x32_f16(
                false, a0, false, bf0, (short)0, acc0, false, false);
            v16h a1 = *reinterpret_cast<const v16h*>(&h_frag[((t + 1) * 32 + l) * 16]);
            v16h bf1 = *reinterpret_cast<const v16h*>(&w_frag[((t + 1) * 2 + g) * 16]);
            acc1 = __builtin_amdgcn_wmma_f32_16x16x32_f16(
                false, a1, false, bf1, (short)0, acc1, false, false);
        }
        const v8f tot = acc0 + acc1;
        const float bb = bias[0];
        // D layout: VGPR r, lanes 0-15 -> D[M=r, N=lane]; lanes 16-31 -> D[M=8+r].
        // Column N==0 lives in lane 0 (rows 0..7) and lane 16 (rows 8..15).
        if (l == 0) {
#pragma unroll
            for (int r = 0; r < 8; ++r) {
                const float z = tot[r] + bb;
                out[b0 + r] = 1.0f / (1.0f + expf(-z));
            }
        } else if (l == 16) {
#pragma unroll
            for (int r = 0; r < 8; ++r) {
                const float z = tot[r] + bb;
                out[b0 + 8 + r] = 1.0f / (1.0f + expf(-z));
            }
        }
    }
}

// ---------------------------------------------------------------------------
extern "C" void kernel_launch(void* const* d_in, const int* in_sizes, int n_in,
                              void* d_out, int out_size, void* d_ws, size_t ws_size,
                              hipStream_t stream) {
    const float* x    = (const float*)d_in[0];
    const float* ln_w = (const float*)d_in[1];
    const float* ln_b = (const float*)d_in[2];
    const float* W    = (const float*)d_in[3];
    const float* bias = (const float*)d_in[4];
    float* out = (float*)d_out;

    const int C   = in_sizes[1];          // 1024
    const int OUT = in_sizes[4];          // 1
    const int B   = out_size / OUT;       // 1024
    const int N   = in_sizes[0] / (B * C);// 256

    float* s = (float*)d_ws;              // B*C*4 = 4 MB scratch

    reduce_n_kernel<<<dim3(B), dim3(C / 4), 0, stream>>>(x, s, N, C);
    ln_head_kernel<<<dim3(B / 16), dim3(256), 0, stream>>>(
        s, ln_w, ln_b, W, bias, out, C);
}
